// GraphSAGE_75350906241117
// MI455X (gfx1250) — matrix operand order repaired and checked by reading, not compiled, not run on hardware
//
#include <hip/hip_runtime.h>

typedef __attribute__((ext_vector_type(16))) __bf16 v16bf;
typedef __attribute__((ext_vector_type(8)))  __bf16 v8bf;
typedef __attribute__((ext_vector_type(8)))  float  v8f;

// ---------------- problem constants (match reference) ----------------
constexpr int N_SRC0 = 100000, N_DST0 = 20000, N_DST1 = 5000;
constexpr int E0 = 160000, E1 = 40000;
constexpr int IN_F = 512, H_F = 1024, N_CLS = 256;

// ---------------- workspace layout (bytes) ----------------
// Aliasing is safe because the stream serializes kernels:
//   OFF_SUM: summed1 f32 (20000x512) -> later summed2 f32 (5000x1024)
//   OFF_A  : A1 bf16 (20000x1024)    -> later A2 bf16 (5000x2048)
constexpr size_t OFF_SUM  = 0;
constexpr size_t OFF_A    = 40960000;
constexpr size_t OFF_H    = 81920000;   // bf16 h (20000x1024)
constexpr size_t OFF_DEG1 = 122880000;  // f32 20000
constexpr size_t OFF_DEG2 = 122960128;  // f32 5000
constexpr size_t OFF_W1   = 122980608;  // bf16 W1^T (1024 x 1024)  rows=N, cols=K
constexpr size_t OFF_W2   = 125077760;  // bf16 W2^T (256 x 2048)
// total ~126.2 MB

// ---------------- utility kernels ----------------
__global__ void zero_f32_kernel(float* __restrict__ p, int n) {
  int i = blockIdx.x * blockDim.x + threadIdx.x;
  if (i < n) p[i] = 0.0f;
}

// Convert f32 (K_src x N, row-major) into transposed bf16 block of W^T:
// dst[n * ldDst + koff + k] = (bf16) src[k * N + n]
__global__ void cvt_t_kernel(__bf16* __restrict__ dst, const float* __restrict__ src,
                             int K_src, int N, int ldDst, int koff) {
  int i = blockIdx.x * blockDim.x + threadIdx.x;
  if (i >= K_src * N) return;
  int k = i / N;
  int n = i - k * N;
  dst[(size_t)n * ldDst + koff + k] = (__bf16)src[i];
}

// ---------------- layer-1 edge scatter: summed1 += x[src], deg1 += 1 ----------------
__global__ __launch_bounds__(256)
void scatter1_kernel(const float* __restrict__ x,
                     const int* __restrict__ esrc, const int* __restrict__ edst,
                     float* __restrict__ summed, float* __restrict__ deg) {
  int tid = blockIdx.x * blockDim.x + threadIdx.x;   // E0 * 128 threads
  int e = tid >> 7;
  int t = tid & 127;
  int s = esrc[e];
  int d = edst[e];
  const float4 v = *(const float4*)(x + (size_t)s * IN_F + t * 4);
  float* out = summed + (size_t)d * IN_F + t * 4;
  atomicAdd(out + 0, v.x);
  atomicAdd(out + 1, v.y);
  atomicAdd(out + 2, v.z);
  atomicAdd(out + 3, v.w);
  if (t == 0) atomicAdd(deg + d, 1.0f);
}

// ---------------- layer-2 edge scatter: summed2 += h[src] (bf16 src, f32 accum) ----------------
__global__ __launch_bounds__(256)
void scatter2_kernel(const __bf16* __restrict__ h,
                     const int* __restrict__ esrc, const int* __restrict__ edst,
                     float* __restrict__ summed, float* __restrict__ deg) {
  int tid = blockIdx.x * blockDim.x + threadIdx.x;   // E1 * 256 threads
  int e = tid >> 8;
  int t = tid & 255;
  int s = esrc[e];
  int d = edst[e];
  const __bf16* hp = h + (size_t)s * H_F + t * 4;
  float* out = summed + (size_t)d * H_F + t * 4;
  atomicAdd(out + 0, (float)hp[0]);
  atomicAdd(out + 1, (float)hp[1]);
  atomicAdd(out + 2, (float)hp[2]);
  atomicAdd(out + 3, (float)hp[3]);
  if (t == 0) atomicAdd(deg + d, 1.0f);
}

// ---------------- build A1 = [x[:20000] | summed1/max(deg1,1)] in bf16 ----------------
__global__ __launch_bounds__(256)
void buildA1_kernel(const float* __restrict__ x, const float* __restrict__ summed,
                    const float* __restrict__ deg, __bf16* __restrict__ A) {
  int i = blockIdx.x * blockDim.x + threadIdx.x;     // 20000*1024 threads
  int row = i >> 10;
  int col = i & 1023;
  float v;
  if (col < IN_F) {
    v = x[(size_t)row * IN_F + col];
  } else {
    float dgv = deg[row];
    v = summed[(size_t)row * IN_F + (col - IN_F)] / (dgv > 1.0f ? dgv : 1.0f);
  }
  A[(size_t)row * (2 * IN_F) + col] = (__bf16)v;
}

// ---------------- build A2 = [h[:5000] | summed2/max(deg2,1)] in bf16 ----------------
__global__ __launch_bounds__(256)
void buildA2_kernel(const __bf16* __restrict__ h, const float* __restrict__ summed,
                    const float* __restrict__ deg, __bf16* __restrict__ A) {
  int i = blockIdx.x * blockDim.x + threadIdx.x;     // 5000*2048 threads
  int row = i >> 11;
  int col = i & 2047;
  __bf16 v;
  if (col < H_F) {
    v = h[(size_t)row * H_F + col];
  } else {
    float dgv = deg[row];
    v = (__bf16)(summed[(size_t)row * H_F + (col - H_F)] / (dgv > 1.0f ? dgv : 1.0f));
  }
  A[(size_t)row * (2 * H_F) + col] = v;
}

// ---------------- tiled bf16 WMMA GEMM: C = act(A @ B^T' + bias) ----------------
// A : M x K bf16 row-major.  BT: N x K bf16 row-major (i.e. B transposed).
// Block tile 128x128x64, 256 threads = 8 waves (4x2), wave tile 32x64 (2x4 wmma),
// two 32-K WMMA substeps per stage.
constexpr int BM = 128, BN = 128, BK = 64;
constexpr int LDSS = 72;   // padded bf16 row stride (144 B: 16 B-aligned, conflict-spread)

template <bool RELU, typename OutT>
__global__ __launch_bounds__(256)
void gemm_bias_kernel(const __bf16* __restrict__ A, const __bf16* __restrict__ BT,
                      const float* __restrict__ bias, OutT* __restrict__ C,
                      int M, int N, int K) {
  __shared__ __bf16 As[BM * LDSS];   // [m][k]
  __shared__ __bf16 Bs[BN * LDSS];   // [n][k]

  const int tid  = threadIdx.x;
  const int lane = tid & 31;
  const int wid  = tid >> 5;
  const int waveM = wid >> 1;        // 0..3
  const int waveN = wid & 1;         // 0..1
  const int bn0 = blockIdx.x * BN;
  const int bm0 = blockIdx.y * BM;
  const int lh  = lane >> 4;         // lane half
  const int l15 = lane & 15;

  v8f acc[2][4];
#pragma unroll
  for (int tm = 0; tm < 2; ++tm)
#pragma unroll
    for (int tn = 0; tn < 4; ++tn)
#pragma unroll
      for (int e = 0; e < 8; ++e) acc[tm][tn][e] = 0.0f;

  // staging split: each thread moves 64 B of A and 64 B of B per stage
  const int st_row = tid >> 1;       // 0..127
  const int st_seg = tid & 1;        // 0..1, 32 bf16 each

  for (int k0 = 0; k0 < K; k0 += BK) {
    __syncthreads();
    {   // stage A tile (zero-fill M overrun): 128 x 64 bf16
      int gm = bm0 + st_row;
      uint4 d0 = make_uint4(0u, 0u, 0u, 0u), d1 = d0, d2 = d0, d3 = d0;
      if (gm < M) {
        const uint4* p = (const uint4*)(A + (size_t)gm * K + k0 + st_seg * 32);
        d0 = p[0]; d1 = p[1]; d2 = p[2]; d3 = p[3];
      }
      uint4* q = (uint4*)(As + st_row * LDSS + st_seg * 32);
      q[0] = d0; q[1] = d1; q[2] = d2; q[3] = d3;
    }
    {   // stage B tile from pre-transposed weights: 128 x 64 bf16 (N always in range)
      const uint4* p = (const uint4*)(BT + (size_t)(bn0 + st_row) * K + k0 + st_seg * 32);
      uint4 d0 = p[0], d1 = p[1], d2 = p[2], d3 = p[3];
      uint4* q = (uint4*)(Bs + st_row * LDSS + st_seg * 32);
      q[0] = d0; q[1] = d1; q[2] = d2; q[3] = d3;
    }
    __syncthreads();

#pragma unroll
    for (int kk = 0; kk < 2; ++kk) {   // two 32-K WMMA substeps per stage
      union Frag { v16bf v; v8bf h[2]; };
      Frag af[2], bfr[4];
#pragma unroll
      for (int tm = 0; tm < 2; ++tm) {           // A frag per ISA 16-bit A layout
        int r = waveM * 32 + tm * 16 + l15;
        const __bf16* base = As + r * LDSS + kk * 32;
        af[tm].h[0] = *(const v8bf*)(base + lh * 8);        // K: lh*8..+7
        af[tm].h[1] = *(const v8bf*)(base + 16 + lh * 8);   // K: 16+lh*8..+7
      }
#pragma unroll
      for (int tn = 0; tn < 4; ++tn) {           // B frag: same K order as A frag
        int n = waveN * 64 + tn * 16 + l15;
        const __bf16* bb = Bs + n * LDSS + kk * 32;
        bfr[tn].h[0] = *(const v8bf*)(bb + lh * 8);
        bfr[tn].h[1] = *(const v8bf*)(bb + 16 + lh * 8);
      }
#pragma unroll
      for (int tm = 0; tm < 2; ++tm)
#pragma unroll
        for (int tn = 0; tn < 4; ++tn)
          acc[tm][tn] = __builtin_amdgcn_wmma_f32_16x16x32_bf16(
              false, af[tm].v, false, bfr[tn].v, (short)0, acc[tm][tn], false, false);
    }
  }

  // epilogue: bias (+ReLU) and store
#pragma unroll
  for (int tm = 0; tm < 2; ++tm) {
#pragma unroll
    for (int tn = 0; tn < 4; ++tn) {
      int col = bn0 + waveN * 64 + tn * 16 + l15;
      float bv = bias[col];
#pragma unroll
      for (int r = 0; r < 8; ++r) {
        int row = bm0 + waveM * 32 + tm * 16 + r + lh * 8;
        if (row < M) {
          float v = acc[tm][tn][r] + bv;
          if (RELU) v = v > 0.0f ? v : 0.0f;
          C[(size_t)row * N + col] = (OutT)v;
        }
      }
    }
  }
}

// ---------------- launch ----------------
extern "C" void kernel_launch(void* const* d_in, const int* in_sizes, int n_in,
                              void* d_out, int out_size, void* d_ws, size_t ws_size,
                              hipStream_t stream) {
  (void)in_sizes; (void)n_in; (void)out_size; (void)ws_size;
  const float* x        = (const float*)d_in[0];
  const float* W_self1  = (const float*)d_in[1];
  const float* W_neigh1 = (const float*)d_in[2];
  const float* b1       = (const float*)d_in[3];
  const float* W_self2  = (const float*)d_in[4];
  const float* W_neigh2 = (const float*)d_in[5];
  const float* b2       = (const float*)d_in[6];
  const int* e0_src = (const int*)d_in[7];
  const int* e0_dst = (const int*)d_in[8];
  const int* e1_src = (const int*)d_in[9];
  const int* e1_dst = (const int*)d_in[10];

  char* ws = (char*)d_ws;
  float*  sum1 = (float*)(ws + OFF_SUM);          // 20000x512
  float*  sum2 = (float*)(ws + OFF_SUM);          // 5000x1024 (alias, re-zeroed)
  __bf16* A1   = (__bf16*)(ws + OFF_A);           // 20000x1024
  __bf16* A2   = (__bf16*)(ws + OFF_A);           // 5000x2048 (alias)
  __bf16* hbuf = (__bf16*)(ws + OFF_H);           // 20000x1024
  float*  deg1 = (float*)(ws + OFF_DEG1);
  float*  deg2 = (float*)(ws + OFF_DEG2);
  __bf16* W1T  = (__bf16*)(ws + OFF_W1);          // 1024 x 1024 (N x K)
  __bf16* W2T  = (__bf16*)(ws + OFF_W2);          // 256 x 2048 (N x K)
  float*  out  = (float*)d_out;                   // 5000x256

  const int T = 256;
  auto blks = [](long long n, int t) { return (unsigned)((n + t - 1) / t); };

  // ---- layer 1 ----
  zero_f32_kernel<<<blks((long long)N_DST0 * IN_F, T), T, 0, stream>>>(sum1, N_DST0 * IN_F);
  zero_f32_kernel<<<blks(N_DST0, T), T, 0, stream>>>(deg1, N_DST0);

  // W1^T rows = H_F, cols = 2*IN_F : [W_self1; W_neigh1]^T
  cvt_t_kernel<<<blks((long long)IN_F * H_F, T), T, 0, stream>>>(W1T, W_self1, IN_F, H_F, 2 * IN_F, 0);
  cvt_t_kernel<<<blks((long long)IN_F * H_F, T), T, 0, stream>>>(W1T, W_neigh1, IN_F, H_F, 2 * IN_F, IN_F);
  // W2^T rows = N_CLS, cols = 2*H_F : [W_self2; W_neigh2]^T
  cvt_t_kernel<<<blks((long long)H_F * N_CLS, T), T, 0, stream>>>(W2T, W_self2, H_F, N_CLS, 2 * H_F, 0);
  cvt_t_kernel<<<blks((long long)H_F * N_CLS, T), T, 0, stream>>>(W2T, W_neigh2, H_F, N_CLS, 2 * H_F, H_F);

  scatter1_kernel<<<blks((long long)E0 * 128, T), T, 0, stream>>>(x, e0_src, e0_dst, sum1, deg1);
  buildA1_kernel<<<blks((long long)N_DST0 * (2 * IN_F), T), T, 0, stream>>>(x, sum1, deg1, A1);

  {
    dim3 grid(H_F / BN, (N_DST0 + BM - 1) / BM);  // (8, 157)
    gemm_bias_kernel<true, __bf16><<<grid, T, 0, stream>>>(A1, W1T, b1, hbuf,
                                                           N_DST0, H_F, 2 * IN_F);
  }

  // ---- layer 2 ----
  zero_f32_kernel<<<blks((long long)N_DST1 * H_F, T), T, 0, stream>>>(sum2, N_DST1 * H_F);
  zero_f32_kernel<<<blks(N_DST1, T), T, 0, stream>>>(deg2, N_DST1);

  scatter2_kernel<<<blks((long long)E1 * 256, T), T, 0, stream>>>(hbuf, e1_src, e1_dst, sum2, deg2);
  buildA2_kernel<<<blks((long long)N_DST1 * (2 * H_F), T), T, 0, stream>>>(hbuf, sum2, deg2, A2);

  {
    dim3 grid(N_CLS / BN, (N_DST1 + BM - 1) / BM);  // (2, 40)
    gemm_bias_kernel<false, float><<<grid, T, 0, stream>>>(A2, W2T, b2, out,
                                                           N_DST1, N_CLS, 2 * H_F);
  }
}
